// TransformerDecoderLayer_37735582663056
// MI455X (gfx1250) — matrix operand order repaired and checked by reading, not compiled, not run on hardware
//
#include <hip/hip_runtime.h>
#include <math.h>

typedef __attribute__((ext_vector_type(16))) _Float16 v16h;
typedef __attribute__((ext_vector_type(8)))  float    v8f;

#define D_    256
#define NH_   8
#define NL_   4
#define NPT_  25
#define HD_   32
#define B_    2
#define LQ_   1000
#define LV_   44965
#define DFF_  1024

// ---------------------------------------------------------------------------
// WMMA GEMM:  O[m,n] = act( sum_k A[m,k] * W[n,k] + bias[n] )
// A: (M,Kd) row-major, W: (N,Kd) row-major (i.e. x @ W^T).
// Batched via blockIdx.z = zo*z2+zi with independent strides.
// One wave computes a 16 x (16*NS) strip; NS is a template constant so the
// K-loop is a single straight-line basic block (no per-iteration branches,
// no PHI copies).  Out-of-range rows/columns use *clamped addresses only*:
// a WMMA output column n depends only on B column n (and row m only on A
// row m), so garbage rows/cols are simply never stored -- no zero-selects
// needed in the fast loop.  Only the K-tail needs k<Kd selects.
// EXEC is all-ones at every v_wmma (all lane conditions are selects).
// ---------------------------------------------------------------------------
struct GemmP {
  const float* A; const float* W; const float* bias; float* O;
  int M, N, Kd, ldO, z2;
  long sAo, sAi, sWo, sWi, sBo, sBi, sOo, sOi;
};

__device__ __forceinline__ void cvt16(const float4& x0, const float4& x1,
                                      const float4& x2, const float4& x3,
                                      v16h& d) {
  d[0]=(_Float16)x0.x; d[1]=(_Float16)x0.y; d[2]=(_Float16)x0.z; d[3]=(_Float16)x0.w;
  d[4]=(_Float16)x1.x; d[5]=(_Float16)x1.y; d[6]=(_Float16)x1.z; d[7]=(_Float16)x1.w;
  d[8]=(_Float16)x2.x; d[9]=(_Float16)x2.y; d[10]=(_Float16)x2.z; d[11]=(_Float16)x2.w;
  d[12]=(_Float16)x3.x; d[13]=(_Float16)x3.y; d[14]=(_Float16)x3.z; d[15]=(_Float16)x3.w;
}

template <int NS, bool RELU>
__global__ __launch_bounds__(128) void gemm_wmma_k(GemmP p) {
  const int lane = threadIdx.x & 31;
  const int wv   = threadIdx.x >> 5;
  const int m0   = (blockIdx.y * 4 + wv) * 16;
  if (m0 >= p.M) return;                     // scalar: uniform branch
  const int n0 = blockIdx.x * (16 * NS);
  const int zo = (int)blockIdx.z / p.z2, zi = (int)blockIdx.z % p.z2;
  const float* A  = p.A + zo * p.sAo + zi * p.sAi;
  const float* W  = p.W + zo * p.sWo + zi * p.sWi;
  const float* Bp = p.bias ? (p.bias + zo * p.sBo + zi * p.sBi) : (const float*)0;
  float*       O  = p.O + zo * p.sOo + zi * p.sOi;

  const int  l15   = lane & 15;
  const int  lhalf = (lane >> 4) << 3;       // A half-select: 0 or 8
  const int  kB    = (lane >> 4) << 4;       // unused (0 or 16); B frag also uses lhalf
  const int  mrow  = m0 + l15;
  const bool mfull = (m0 + 16 <= p.M);
  const float* arow = A + (long)min(mrow, p.M - 1) * p.Kd;        // clamped

  const float* wrow[NS];
#pragma unroll
  for (int s = 0; s < NS; ++s)
    wrow[s] = W + (long)min(n0 + s * 16 + l15, p.N - 1) * p.Kd;   // clamped

  v8f acc[NS];
#pragma unroll
  for (int s = 0; s < NS; ++s) {
    float bv = Bp ? Bp[min(n0 + s * 16 + l15, p.N - 1)] : 0.f;
#pragma unroll
    for (int r = 0; r < 8; ++r) acc[s][r] = bv;
  }

  const int kfull = p.Kd & ~31;

  // ---- fast path: straight-line body, unconditional b128 loads -----------
  for (int kk = 0; kk < kfull; kk += 32) {
    // A frag: h0..7 -> K=kk+lhalf+0..7, h8..15 -> K=kk+16+lhalf+0..7
    const float* ap = arow + kk + lhalf;
    float4 a0 = *(const float4*)(ap + 0);
    float4 a1 = *(const float4*)(ap + 4);
    float4 a2 = *(const float4*)(ap + 16);
    float4 a3 = *(const float4*)(ap + 20);
    v16h a; cvt16(a0, a1, a2, a3, a);
#pragma unroll
    for (int s = 0; s < NS; ++s) {
      // B frag: h0..7 -> K=kk+lhalf+0..7, h8..15 -> K=kk+16+lhalf+0..7
      const float* wp = wrow[s] + kk + lhalf;
      float4 b0 = *(const float4*)(wp + 0);
      float4 b1 = *(const float4*)(wp + 4);
      float4 b2 = *(const float4*)(wp + 16);
      float4 b3 = *(const float4*)(wp + 20);
      v16h bm; cvt16(b0, b1, b2, b3, bm);
      acc[s] = __builtin_amdgcn_wmma_f32_16x16x32_f16(
          false, a, false, bm, (short)0, acc[s], false, false);
    }
  }

  // ---- K tail (Kd % 32 != 0): clamp addresses, zero-select on k only -----
  if (kfull < p.Kd) {
    v16h a;
#pragma unroll
    for (int h = 0; h < 16; ++h) {
      int k  = kfull + ((h >> 3) << 4) + lhalf + (h & 7);
      float v = arow[min(k, p.Kd - 1)];
      a[h] = (k < p.Kd) ? (_Float16)v : (_Float16)0.f;
    }
#pragma unroll
    for (int s = 0; s < NS; ++s) {
      v16h bm;
#pragma unroll
      for (int h = 0; h < 16; ++h) {
        int k  = kfull + ((h >> 3) << 4) + lhalf + (h & 7);
        float v = wrow[s][min(k, p.Kd - 1)];
        bm[h] = (k < p.Kd) ? (_Float16)v : (_Float16)0.f;
      }
      acc[s] = __builtin_amdgcn_wmma_f32_16x16x32_f16(
          false, a, false, bm, (short)0, acc[s], false, false);
    }
  }

  // ---- store: C/D layout lane L, VGPR r -> (m = m0+8*(L/16)+r, n) --------
#pragma unroll
  for (int s = 0; s < NS; ++s) {
    if (n0 + s * 16 >= p.N) continue;        // scalar: uniform branch
    const int  n  = n0 + s * 16 + l15;
    const bool nf = (n0 + s * 16 + 16 <= p.N);
    if (mfull && nf) {
#pragma unroll
      for (int r = 0; r < 8; ++r) {
        float v = acc[s][r];
        if (RELU) v = fmaxf(v, 0.f);
        O[(long)(m0 + lhalf + r) * p.ldO + n] = v;
      }
    } else {
#pragma unroll
      for (int r = 0; r < 8; ++r) {
        int m = m0 + lhalf + r;
        if (m < p.M && n < p.N) {
          float v = acc[s][r];
          if (RELU) v = fmaxf(v, 0.f);
          O[(long)m * p.ldO + n] = v;
        }
      }
    }
  }
}

// ---------------------------------------------------------------------------
__global__ void add_k(const float* a, const float* b, float* o, int n) {
  int i = blockIdx.x * blockDim.x + threadIdx.x;
  if (i < n) o[i] = a[i] + b[i];
}

// V transpose: in [z][LQ][HD] -> out [z][HD][LQ]  (z = B*NH)
__global__ void transpose_v_k(const float* __restrict__ in,
                              float* __restrict__ out) {
  long i = (long)blockIdx.x * blockDim.x + threadIdx.x;
  if (i >= (long)B_ * NH_ * LQ_ * HD_) return;
  long z = i / (LQ_ * HD_), r = i % (LQ_ * HD_);
  long n = r / LQ_, k = r % LQ_;
  out[i] = in[z * (LQ_ * HD_) + k * HD_ + n];
}

// ---------------------------------------------------------------------------
// Row softmax with pre-scale; row r at base + (r/inner)*ostride +
// (r%inner)*istride, contiguous length len.
// ---------------------------------------------------------------------------
__global__ __launch_bounds__(256) void softmax_k(float* base, int inner,
                                                 long ostride, long istride,
                                                 int len, float scale) {
  long r = blockIdx.x;
  float* p = base + (r / inner) * ostride + (r % inner) * istride;
  __shared__ float red[256];
  int t = threadIdx.x;
  float mx = -3.402823466e38f;
  for (int i = t; i < len; i += 256) mx = fmaxf(mx, p[i] * scale);
  red[t] = mx; __syncthreads();
  for (int s = 128; s > 0; s >>= 1) {
    if (t < s) red[t] = fmaxf(red[t], red[t + s]);
    __syncthreads();
  }
  mx = red[0]; __syncthreads();
  float sum = 0.f;
  for (int i = t; i < len; i += 256) {
    float e = expf(p[i] * scale - mx);
    p[i] = e; sum += e;
  }
  red[t] = sum; __syncthreads();
  for (int s = 128; s > 0; s >>= 1) {
    if (t < s) red[t] += red[t + s];
    __syncthreads();
  }
  float inv = 1.f / red[0];
  for (int i = t; i < len; i += 256) p[i] *= inv;
}

// ---------------------------------------------------------------------------
// out_row = LayerNorm(a_row + b_row) * g + be   (rows of 256)
// ---------------------------------------------------------------------------
__global__ __launch_bounds__(256) void add_ln_k(const float* a, const float* b,
                                                const float* g, const float* be,
                                                float* out) {
  long r = blockIdx.x;
  int t = threadIdx.x;
  float v = a[r * D_ + t] + b[r * D_ + t];
  __shared__ float red[256];
  red[t] = v; __syncthreads();
  for (int s = 128; s > 0; s >>= 1) {
    if (t < s) red[t] += red[t + s];
    __syncthreads();
  }
  float mu = red[0] * (1.f / D_); __syncthreads();
  float d = v - mu;
  red[t] = d * d; __syncthreads();
  for (int s = 128; s > 0; s >>= 1) {
    if (t < s) red[t] += red[t + s];
    __syncthreads();
  }
  float var = red[0] * (1.f / D_);
  out[r * D_ + t] = d * rsqrtf(var + 1e-5f) * g[t] + be[t];
}

// ---------------------------------------------------------------------------
// Rotated-box deformable sampling; one wave per (b,q,h), lane = head dim.
// Branch-free: clamped gather, validity folded into the weight (matches the
// reference, which clips indices and multiplies the weight by `valid`).
// ---------------------------------------------------------------------------
__global__ __launch_bounds__(128) void box_sample_k(
    const float* __restrict__ value, const float* __restrict__ aw,
    const float* __restrict__ ob, const float* __restrict__ refw,
    const int* __restrict__ mshape, const int* __restrict__ mstart,
    float* __restrict__ out) {
  int unit = blockIdx.x * 4 + (threadIdx.x >> 5);   // 0 .. B*LQ*NH-1
  if (unit >= B_ * LQ_ * NH_) return;               // wave-uniform
  int lane = threadIdx.x & 31;
  int h  = unit % NH_;
  int bq = unit / NH_;                              // b*LQ + q
  int b  = bq / LQ_;
  const float* rw = refw + (long)bq * 8;
  float cxr = rw[0], cyr = rw[1], wr = rw[3], hr = rw[4], ar = rw[6];
  const float* obp = ob + (long)bq * 160 + h * 20;
  const float* awp = aw + (long)bq * 800 + h * 100;
  const float* vb = value + (long)b * LV_ * D_ + h * HD_ + lane;
  float acc = 0.f;
  for (int l = 0; l < NL_; ++l) {
    float offx = obp[l * 5 + 0], offy = obp[l * 5 + 1];
    float offw = obp[l * 5 + 2], offh = obp[l * 5 + 3];
    float offa = obp[l * 5 + 4];
    float ang = (ar + offa * (1.f / 16.f)) * 6.283185307179586f;
    float bx = cxr + offx * 0.125f * wr;
    float by = cyr + offy * 0.125f * hr;
    float sx = fmaxf(wr + offw * 0.125f * wr, 0.f);
    float sy = fmaxf(hr + offh * 0.125f * hr, 0.f);
    float cs = cosf(ang), sn = sinf(ang);
    int H = mshape[l * 2 + 0], W = mshape[l * 2 + 1];
    int st = mstart[l];
    for (int p = 0; p < NPT_; ++p) {
      float gx = (float)(p % 5 - 2) * 0.2f * sx;
      float gy = (float)(p / 5 - 2) * 0.2f * sy;
      float lx = bx + cs * gx - sn * gy;
      float ly = by + sn * gx + cs * gy;
      float x = lx * (float)W - 0.5f;
      float y = ly * (float)H - 0.5f;
      float x0 = floorf(x), y0 = floorf(y);
      float a = awp[l * NPT_ + p];
#pragma unroll
      for (int dy = 0; dy < 2; ++dy)
#pragma unroll
        for (int dx = 0; dx < 2; ++dx) {
          float xi = x0 + dx, yi = y0 + dy;
          float ok = (xi >= 0.f && xi < (float)W &&
                      yi >= 0.f && yi < (float)H) ? 1.f : 0.f;
          int xc = min(max((int)xi, 0), W - 1);
          int yc = min(max((int)yi, 0), H - 1);
          float wgt = (1.f - fabsf(x - xi)) * (1.f - fabsf(y - yi)) * ok * a;
          long fi = st + (long)yc * W + xc;
          acc += wgt * vb[fi * D_];                 // unconditional load
        }
    }
  }
  out[(long)bq * D_ + h * HD_ + lane] = acc;
}

// ---------------------------------------------------------------------------
// Host-side orchestration
// ---------------------------------------------------------------------------
extern "C" void kernel_launch(void* const* d_in, const int* in_sizes, int n_in,
                              void* d_out, int out_size, void* d_ws,
                              size_t ws_size, hipStream_t stream) {
  (void)in_sizes; (void)n_in; (void)out_size; (void)ws_size;
  const float* query    = (const float*)d_in[1];
  const float* refw     = (const float*)d_in[2];
  const float* memory   = (const float*)d_in[3];
  const int*   mshape   = (const int*)d_in[4];
  const int*   mstart   = (const int*)d_in[5];
  const float* sa_in_w  = (const float*)d_in[6];
  const float* sa_in_b  = (const float*)d_in[7];
  const float* sa_out_w = (const float*)d_in[8];
  const float* sa_out_b = (const float*)d_in[9];
  const float* pos_w1   = (const float*)d_in[10];
  const float* pos_b1   = (const float*)d_in[11];
  const float* pos_w2   = (const float*)d_in[12];
  const float* pos_b2   = (const float*)d_in[13];
  const float* pos_w3   = (const float*)d_in[14];
  const float* pos_b3   = (const float*)d_in[15];
  const float* box_w    = (const float*)d_in[16];
  const float* box_b    = (const float*)d_in[17];
  const float* battn_w  = (const float*)d_in[18];
  const float* battn_b  = (const float*)d_in[19];
  const float* vproj_w  = (const float*)d_in[20];
  const float* vproj_b  = (const float*)d_in[21];
  const float* boxout_w = (const float*)d_in[22];
  const float* boxout_b = (const float*)d_in[23];
  const float* ffn_w1   = (const float*)d_in[24];
  const float* ffn_b1   = (const float*)d_in[25];
  const float* ffn_w2   = (const float*)d_in[26];
  const float* ffn_b2   = (const float*)d_in[27];
  const float* n1_g     = (const float*)d_in[28];
  const float* n1_b     = (const float*)d_in[29];
  const float* n2_g     = (const float*)d_in[30];
  const float* n2_b     = (const float*)d_in[31];
  const float* n3_g     = (const float*)d_in[32];
  const float* n3_b     = (const float*)d_in[33];

  float* ws = (float*)d_ws;
  size_t off = 0;
  auto alloc = [&](size_t n) { float* p = ws + off; off += n; return p; };
  const size_t NQ = (size_t)B_ * LQ_;                  // 2000
  float* qp  = alloc(NQ * D_);
  float* h1  = alloc(NQ * D_);
  float* h2  = alloc(NQ * D_);
  float* qk  = alloc(NQ * D_);
  float* qb  = alloc(NQ * D_);                         // Q [B,NH,LQ,HD]
  float* kb  = alloc(NQ * D_);                         // K
  float* vbf = alloc(NQ * D_);                         // V [B,NH,LQ,HD]
  float* vt  = alloc(NQ * D_);                         // V^T [B,NH,HD,LQ]
  float* att = alloc(NQ * D_);                         // attn out [B,LQ,D]
  float* q2  = alloc(NQ * D_);
  float* x1  = alloc(NQ * D_);
  float* x2  = alloc(NQ * D_);
  float* x3  = alloc(NQ * D_);
  float* awb = alloc(NQ * (size_t)(NH_ * NL_ * NPT_)); // 2000 x 800
  float* obb = alloc(NQ * (size_t)(NH_ * NL_ * 5));    // 2000 x 160
  float* smp = alloc(NQ * D_);
  float* ffh = alloc(NQ * DFF_);
  float* val = alloc((size_t)B_ * LV_ * D_);           // 92 MB
  float* sc  = alloc((size_t)B_ * NH_ * LQ_ * LQ_);    // 64 MB

  auto gemm = [&](const float* A, const float* W, const float* bias, float* O,
                  int M, int N, int Kd, int ldO, bool relu,
                  int Z = 1, int z2 = 1, long sAo = 0, long sAi = 0,
                  long sWo = 0, long sWi = 0, long sBo = 0, long sBi = 0,
                  long sOo = 0, long sOi = 0) {
    GemmP p{A, W, bias, O, M, N, Kd, ldO, z2,
            sAo, sAi, sWo, sWi, sBo, sBi, sOo, sOi};
    int ns = (N >= 64) ? 4 : (N >= 32) ? 2 : 1;
    dim3 g((unsigned)((N + 16 * ns - 1) / (16 * ns)),
           (unsigned)((M + 63) / 64), (unsigned)Z);
    dim3 b(128);
    switch (ns * 2 + (relu ? 1 : 0)) {
      case 9: gemm_wmma_k<4, true ><<<g, b, 0, stream>>>(p); break;
      case 8: gemm_wmma_k<4, false><<<g, b, 0, stream>>>(p); break;
      case 5: gemm_wmma_k<2, true ><<<g, b, 0, stream>>>(p); break;
      case 4: gemm_wmma_k<2, false><<<g, b, 0, stream>>>(p); break;
      case 3: gemm_wmma_k<1, true ><<<g, b, 0, stream>>>(p); break;
      default: gemm_wmma_k<1, false><<<g, b, 0, stream>>>(p); break;
    }
  };

  const int NE = (int)(NQ * D_);   // 512000

  // ---- positional MLP -----------------------------------------------------
  gemm(refw, pos_w1, pos_b1, h1, (int)NQ, D_, 8, D_, true);
  gemm(h1, pos_w2, pos_b2, h2, (int)NQ, D_, D_, D_, true);
  gemm(h2, pos_w3, pos_b3, qp, (int)NQ, D_, D_, D_, false);
  add_k<<<(NE + 255) / 256, 256, 0, stream>>>(query, qp, qk, NE);

  // ---- self-attention -----------------------------------------------------
  const long sA_b = (long)LQ_ * D_;
  const long sW_h = (long)HD_ * D_;
  const long sO_b = (long)NH_ * LQ_ * HD_;
  const long sO_h = (long)LQ_ * HD_;
  gemm(qk, sa_in_w, sa_in_b, qb, LQ_, HD_, D_, HD_, false,
       B_ * NH_, NH_, sA_b, 0, 0, sW_h, 0, HD_, sO_b, sO_h);
  gemm(qk, sa_in_w + D_ * D_, sa_in_b + D_, kb, LQ_, HD_, D_, HD_, false,
       B_ * NH_, NH_, sA_b, 0, 0, sW_h, 0, HD_, sO_b, sO_h);
  gemm(query, sa_in_w + 2 * D_ * D_, sa_in_b + 2 * D_, vbf, LQ_, HD_, D_, HD_,
       false, B_ * NH_, NH_, sA_b, 0, 0, sW_h, 0, HD_, sO_b, sO_h);
  // scores[z] = Q[z] @ K[z]^T
  gemm(qb, kb, nullptr, sc, LQ_, LQ_, HD_, LQ_, false,
       B_ * NH_, 1, sO_h, 0, sO_h, 0, 0, 0, (long)LQ_ * LQ_, 0);
  softmax_k<<<B_ * NH_ * LQ_, 256, 0, stream>>>(
      sc, 1, (long)LQ_, 0, LQ_, 0.17677669529663687f);   // 1/sqrt(32)
  // V^T so P @ V uses the contiguous weight path
  transpose_v_k<<<(NE + 255) / 256, 256, 0, stream>>>(vbf, vt);
  gemm(sc, vt, nullptr, att, LQ_, HD_, LQ_, D_, false,
       B_ * NH_, NH_, (long)NH_ * LQ_ * LQ_, (long)LQ_ * LQ_,
       sO_b, sO_h, 0, 0, (long)LQ_ * D_, (long)HD_);
  gemm(att, sa_out_w, sa_out_b, q2, (int)NQ, D_, D_, D_, false);
  add_ln_k<<<(unsigned)NQ, 256, 0, stream>>>(query, q2, n1_g, n1_b, x1);

  // ---- box attention ------------------------------------------------------
  add_k<<<(NE + 255) / 256, 256, 0, stream>>>(x1, qp, x2, NE);
  gemm(memory, vproj_w, vproj_b, val, B_ * LV_, D_, D_, D_, false);
  gemm(x2, battn_w, battn_b, awb, (int)NQ, NH_ * NL_ * NPT_, D_,
       NH_ * NL_ * NPT_, false);
  softmax_k<<<B_ * LQ_ * NH_, 256, 0, stream>>>(
      awb, NH_, (long)(NH_ * NL_ * NPT_), (long)(NL_ * NPT_), NL_ * NPT_, 1.f);
  gemm(x2, box_w, box_b, obb, (int)NQ, NH_ * NL_ * 5, D_, NH_ * NL_ * 5, false);
  box_sample_k<<<(B_ * LQ_ * NH_) / 4, 128, 0, stream>>>(
      val, awb, obb, refw, mshape, mstart, smp);
  gemm(smp, boxout_w, boxout_b, q2, (int)NQ, D_, D_, D_, false);
  add_ln_k<<<(unsigned)NQ, 256, 0, stream>>>(x1, q2, n2_g, n2_b, x3);

  // ---- FFN ----------------------------------------------------------------
  gemm(x3, ffn_w1, ffn_b1, ffh, (int)NQ, DFF_, D_, DFF_, true);
  gemm(ffh, ffn_w2, ffn_b2, q2, (int)NQ, D_, DFF_, D_, false);
  add_ln_k<<<(unsigned)NQ, 256, 0, stream>>>(x3, q2, n3_g, n3_b,
                                             (float*)d_out);
}
